// MultiHeadSelfAtt_25915832664785
// MI455X (gfx1250) — matrix operand order repaired and checked by reading, not compiled, not run on hardware
//
#include <hip/hip_runtime.h>
#include <hip/hip_bf16.h>

#define BATCH  2
#define SEQ    2048
#define DMODEL 512
#define NHEAD  8
#define VDIM   64
#define BH     (BATCH * NHEAD)

typedef __attribute__((ext_vector_type(16))) __bf16 bf16x16;
typedef __attribute__((ext_vector_type(8)))  float  f32x8;

union FragAB {
    bf16x16 v;
    __bf16  e[16];
    uint4   q[2];
};

__device__ __forceinline__ __bf16 f2bf(float f) {
    // native fptrunc: backend selects hardware bf16 convert on gfx1250
    return (__bf16)f;
}

union PairBF { unsigned int u; __bf16 b[2]; };

// -------------------------------------------------------------------------
// Kernel 1: QKV projection.  For (b,h,which) compute X[N,D] * W[D,VD] in
// bf16 WMMA with fp32 accumulate.  Q scaled by 1/sqrt(VD).  Q,K stored
// bf16 [bh][tok][64]; V stored transposed bf16 [bh][vd][tok].
// Block = 256 threads = 8 waves; each wave owns one 16x16 output tile.
// blockIdx.x: (rowGroup<<2)|colTile ; blockIdx.y: which ; blockIdx.z: bh
// -------------------------------------------------------------------------
__global__ __launch_bounds__(256) void qkv_proj_kernel(
    const float* __restrict__ x,     // [B][N][D]
    const float* __restrict__ qw,    // [H][D][VD]
    const float* __restrict__ kw,
    const float* __restrict__ vw,
    __bf16* __restrict__ q_ws,       // [BH][N][VD]
    __bf16* __restrict__ k_ws,       // [BH][N][VD]
    __bf16* __restrict__ vt_ws)      // [BH][VD][N]
{
    const int which  = blockIdx.y;           // 0=q 1=k 2=v
    const int bh     = blockIdx.z;           // 0..15
    const int b      = bh >> 3;
    const int lane   = threadIdx.x & 31;
    const int wave   = threadIdx.x >> 5;
    const int colTile = blockIdx.x & 3;                  // 0..3
    const int rowTile = ((blockIdx.x >> 2) << 3) + wave; // 0..127
    const int row0 = rowTile * 16;
    const int col0 = colTile * 16;

    const float* W = ((which == 0) ? qw : (which == 1) ? kw : vw)
                     + (size_t)(bh & 7) * DMODEL * VDIM;

    // W tile staged as bf16 in B-fragment-friendly [n][k] layout (64B rows)
    __shared__ __bf16 ldsB[16][32];

    f32x8 acc = {0.f, 0.f, 0.f, 0.f, 0.f, 0.f, 0.f, 0.f};

    const int m    = lane & 15;
    const int half = lane >> 4;               // 0: lanes 0-15, 1: lanes 16-31
    const float* xrow = x + ((size_t)b * SEQ + row0 + m) * DMODEL;

    for (int k0 = 0; k0 < DMODEL; k0 += 32) {
        // ---- cooperative stage of W[k0..k0+32) x cols[col0..col0+16) ----
        {
            int idx = threadIdx.x * 2;        // 512 bf16 elements
            int n   = idx >> 5;               // 0..15
            int k   = idx & 31;               // even
            PairBF pk;
            pk.b[0] = f2bf(W[(size_t)(k0 + k)     * VDIM + col0 + n]);
            pk.b[1] = f2bf(W[(size_t)(k0 + k + 1) * VDIM + col0 + n]);
            ((unsigned int*)&ldsB[0][0])[(n * 32 + k) >> 1] = pk.u;
        }
        // prefetch next chunk of X and W while this tile is consumed
        if (k0 + 32 < DMODEL) {
            __builtin_prefetch(xrow + k0 + 32, 0, 3);
            __builtin_prefetch(&W[(size_t)(k0 + 32 + (lane & 31)) * VDIM + col0], 0, 3);
        }
        __syncthreads();

        // ---- A fragment: 16x32 bf16, lane=row m, K = 16*half + e (contig) ----
        FragAB a;
        {
            const int kA = k0 + 16 * half;
            float4 a0 = *(const float4*)(xrow + kA);
            float4 a1 = *(const float4*)(xrow + kA + 4);
            float4 a2 = *(const float4*)(xrow + kA + 8);
            float4 a3 = *(const float4*)(xrow + kA + 12);
            a.e[0] = f2bf(a0.x); a.e[1] = f2bf(a0.y);
            a.e[2] = f2bf(a0.z); a.e[3] = f2bf(a0.w);
            a.e[4] = f2bf(a1.x); a.e[5] = f2bf(a1.y);
            a.e[6] = f2bf(a1.z); a.e[7] = f2bf(a1.w);
            a.e[8]  = f2bf(a2.x); a.e[9]  = f2bf(a2.y);
            a.e[10] = f2bf(a2.z); a.e[11] = f2bf(a2.w);
            a.e[12] = f2bf(a3.x); a.e[13] = f2bf(a3.y);
            a.e[14] = f2bf(a3.z); a.e[15] = f2bf(a3.w);
        }

        // ---- B fragment: 32x16 bf16, lane=col n, K = 16*half + e (contig) ----
        FragAB bf;
        {
            const uint4* lrow = (const uint4*)&ldsB[m][16 * half];
            bf.q[0] = lrow[0];
            bf.q[1] = lrow[1];
        }

        acc = __builtin_amdgcn_wmma_f32_16x16x32_bf16(
            false, a.v, false, bf.v, (short)0, acc, false, false);

        __syncthreads();
    }

    // ---- store ----
    if (which < 2) {
        const float sc = (which == 0) ? 0.125f : 1.0f;   // 1/sqrt(64) folded into Q
        __bf16* dst = ((which == 0) ? q_ws : k_ws) + (size_t)bh * SEQ * VDIM;
        const int mBase = 8 * half;
#pragma unroll
        for (int i = 0; i < 8; ++i) {
            int tok = row0 + mBase + i;
            dst[(size_t)tok * VDIM + col0 + m] = f2bf(acc[i] * sc);
        }
    } else {
        // transposed V: [bh][vd][tok], each lane writes 8 contiguous tokens
        union { uint4 q; __bf16 s[8]; } pk;
#pragma unroll
        for (int i = 0; i < 8; ++i) pk.s[i] = f2bf(acc[i]);
        const int vd   = col0 + m;
        const int tok0 = row0 + 8 * half;
        *(uint4*)(vt_ws + ((size_t)bh * VDIM + vd) * SEQ + tok0) = pk.q;
    }
}

// -------------------------------------------------------------------------
// Kernel 2: streaming attention + GELU.
// One wave owns one 16-query tile for one (b,h); loops keys in steps of 32.
// Per step: 4 WMMAs for S = Q*K^T, online softmax (shfl over 16-lane halves
// matching the C-fragment row layout), P reshaped C->A via per-wave LDS,
// 4 WMMAs for O += P*V.  Output: 1/l, exact GELU, [B][N][H*VD] fp32.
// -------------------------------------------------------------------------
__global__ __launch_bounds__(256) void attn_kernel(
    const __bf16* __restrict__ q_ws,   // [BH][N][64]
    const __bf16* __restrict__ k_ws,   // [BH][N][64]
    const __bf16* __restrict__ vt_ws,  // [BH][64][N]
    float* __restrict__ out)           // [B][N][512]
{
    const int lane  = threadIdx.x & 31;
    const int wave  = threadIdx.x >> 5;
    const int wid   = blockIdx.x * 8 + wave;   // 0..2047
    const int bh    = wid >> 7;                // 0..15
    const int qtile = wid & 127;               // 0..127
    const int b     = bh >> 3;
    const int h     = bh & 7;
    const int m     = lane & 15;
    const int half  = lane >> 4;

    __shared__ __bf16 ldsP[8][16][32];  // per-wave P staging (1KB each)

    // ---- Q A-fragments: two 16x32 chunks covering vd 0..63 ----
    FragAB qf[2];
    {
        const __bf16* qr = q_ws + ((size_t)bh * SEQ + qtile * 16 + m) * VDIM;
        const int kb = 16 * half;
#pragma unroll
        for (int f = 0; f < 2; ++f) {
            qf[f].q[0] = *(const uint4*)(qr + f * 32 + kb);
            qf[f].q[1] = *(const uint4*)(qr + f * 32 + kb + 8);
        }
    }

    f32x8 o[4];
#pragma unroll
    for (int j = 0; j < 4; ++j) o[j] = (f32x8){0.f,0.f,0.f,0.f,0.f,0.f,0.f,0.f};
    float mrun[8], lrun[8];
#pragma unroll
    for (int i = 0; i < 8; ++i) { mrun[i] = -1.0e30f; lrun[i] = 0.f; }

    const __bf16* kbase = k_ws + (size_t)bh * SEQ * VDIM;
    const __bf16* vbase = vt_ws + (size_t)bh * VDIM * SEQ;

    for (int key0 = 0; key0 < SEQ; key0 += 32) {
        // prefetch next key step (K rows + V^T columns) into caches
        if (key0 + 32 < SEQ) {
            __builtin_prefetch(kbase + (size_t)(key0 + 32 + lane) * VDIM, 0, 3);
            __builtin_prefetch(vbase + (size_t)(lane * 2) * SEQ + key0 + 32, 0, 3);
        }

        // ---- S = Q * K^T for two 16-key subtiles ----
        f32x8 s[2];
#pragma unroll
        for (int t = 0; t < 2; ++t) {
            const __bf16* kr = kbase + (size_t)(key0 + t * 16 + m) * VDIM;
            const int kb = 16 * half;
            FragAB k0f, k1f;
            k0f.q[0] = *(const uint4*)(kr + kb);
            k0f.q[1] = *(const uint4*)(kr + kb + 8);
            k1f.q[0] = *(const uint4*)(kr + 32 + kb);
            k1f.q[1] = *(const uint4*)(kr + 32 + kb + 8);
            f32x8 sc = (f32x8){0.f,0.f,0.f,0.f,0.f,0.f,0.f,0.f};
            sc = __builtin_amdgcn_wmma_f32_16x16x32_bf16(
                false, qf[0].v, false, k0f.v, (short)0, sc, false, false);
            sc = __builtin_amdgcn_wmma_f32_16x16x32_bf16(
                false, qf[1].v, false, k1f.v, (short)0, sc, false, false);
            s[t] = sc;
        }

        // ---- online softmax; row i lives in vgpr i across 16 lanes of a half ----
#pragma unroll
        for (int i = 0; i < 8; ++i) {
            float rm = fmaxf(s[0][i], s[1][i]);
            rm = fmaxf(rm, __shfl_xor(rm, 1));
            rm = fmaxf(rm, __shfl_xor(rm, 2));
            rm = fmaxf(rm, __shfl_xor(rm, 4));
            rm = fmaxf(rm, __shfl_xor(rm, 8));
            float nm   = fmaxf(mrun[i], rm);
            float corr = __expf(mrun[i] - nm);
            mrun[i] = nm;
            float p0 = __expf(s[0][i] - nm);
            float p1 = __expf(s[1][i] - nm);
            s[0][i] = p0;
            s[1][i] = p1;
            float rs = p0 + p1;
            rs += __shfl_xor(rs, 1);
            rs += __shfl_xor(rs, 2);
            rs += __shfl_xor(rs, 4);
            rs += __shfl_xor(rs, 8);
            lrun[i] = lrun[i] * corr + rs;
#pragma unroll
            for (int j = 0; j < 4; ++j) o[j][i] *= corr;
        }

        // ---- reshape P: C-layout -> LDS -> A-layout (bf16) ----
        {
            const int mb = 8 * half;
#pragma unroll
            for (int i = 0; i < 8; ++i) {
                PairBF pk;
                pk.b[0] = f2bf(s[0][i]);
                pk.b[1] = f2bf(s[1][i]);
                ldsP[wave][mb + i][m]      = pk.b[0];
                ldsP[wave][mb + i][16 + m] = pk.b[1];
            }
        }
        asm volatile("s_wait_dscnt 0x0" ::: "memory");
        FragAB pf;
        {
            const __bf16* prow = &ldsP[wave][m][0];
            const int kb = 16 * half;
            pf.q[0] = *(const uint4*)(prow + kb);
            pf.q[1] = *(const uint4*)(prow + kb + 8);
        }

        // ---- O += P * V  (V^T layout -> contiguous B fragments) ----
#pragma unroll
        for (int j = 0; j < 4; ++j) {
            const __bf16* vr =
                vbase + (size_t)(j * 16 + m) * SEQ + key0 + 16 * half;
            FragAB vf;
            vf.q[0] = *(const uint4*)vr;
            vf.q[1] = *(const uint4*)(vr + 8);
            o[j] = __builtin_amdgcn_wmma_f32_16x16x32_bf16(
                false, pf.v, false, vf.v, (short)0, o[j], false, false);
        }
    }

    // ---- finalize: 1/l, exact GELU, store [B][N][H*VD] ----
    float inv[8];
#pragma unroll
    for (int i = 0; i < 8; ++i) inv[i] = 1.0f / lrun[i];
    const int tokb = qtile * 16 + 8 * half;
#pragma unroll
    for (int j = 0; j < 4; ++j) {
#pragma unroll
        for (int i = 0; i < 8; ++i) {
            float xg = o[j][i] * inv[i];
            xg = 0.5f * xg * (1.0f + erff(xg * 0.70710678118654752f));
            out[((size_t)b * SEQ + tokb + i) * (NHEAD * VDIM) + h * VDIM + j * 16 + m] = xg;
        }
    }
}

// -------------------------------------------------------------------------
extern "C" void kernel_launch(void* const* d_in, const int* in_sizes, int n_in,
                              void* d_out, int out_size, void* d_ws, size_t ws_size,
                              hipStream_t stream) {
    const float* x  = (const float*)d_in[0];
    const float* qw = (const float*)d_in[1];
    const float* kw = (const float*)d_in[2];
    const float* vw = (const float*)d_in[3];
    float* out = (float*)d_out;

    const size_t per = (size_t)BH * SEQ * VDIM;   // elements per bf16 tensor
    __bf16* q_ws  = (__bf16*)d_ws;
    __bf16* k_ws  = q_ws + per;
    __bf16* vt_ws = k_ws + per;

    // 128 row tiles / 8 waves = 16 row groups; x = rowGroup*4 + colTile
    dim3 gProj(64, 3, BH);
    qkv_proj_kernel<<<gProj, 256, 0, stream>>>(x, qw, kw, vw, q_ws, k_ws, vt_ws);

    // 16 bh * 128 qtiles = 2048 waves / 8 per block = 256 blocks
    attn_kernel<<<dim3(256), 256, 0, stream>>>(q_ws, k_ws, vt_ws, out);
}
